// GCN_GW_86835648790601
// MI455X (gfx1250) — matrix operand order repaired and checked by reading, not compiled, not run on hardware
//
#include <hip/hip_runtime.h>

// ---------------------------------------------------------------------------
// CDNA5 (gfx1250) implementation of the GCN + shared-workspace attention block.
// Big GEMMs use v_wmma_f32_16x16x32_bf16 (wave32, 16-row tiles per wave).
// ---------------------------------------------------------------------------

typedef __attribute__((ext_vector_type(16))) __bf16 v16bf;
typedef __attribute__((ext_vector_type(8)))  __bf16 v8bf;
typedef __attribute__((ext_vector_type(8)))  float  v8f;

#define DIV_UP(a,b) (((a)+(b)-1)/(b))

// ---------------- pack xc = [x | pe] as bf16 rows (lda=192) ----------------
__global__ void pack_xc_k(const float* __restrict__ x, const float* __restrict__ pe,
                          __bf16* __restrict__ xc, int n) {
    long long idx = (long long)blockIdx.x * blockDim.x + threadIdx.x;
    if (idx >= (long long)n * 192) return;
    int i = (int)(idx / 192), j = (int)(idx % 192);
    float v = (j < 128) ? x[(size_t)i * 128 + j] : pe[(size_t)i * 64 + (j - 128)];
    xc[idx] = (__bf16)v;
}

// ---- pack a [K,Ncols] row-major fp32 weight into WMMA B-fragment layout ----
// out index = ((kc*NT + nt)*32 + lane)*16 + e ; element (lane,e) holds
// W[kc*32 + 8*(lane>>4) + (e&7) + 16*(e>>3)][nt*16 + (lane&15)]  (zero-padded past Ncols)
__global__ void pack_w_k(const float* __restrict__ W, int Ncols, int NT,
                         int total, __bf16* __restrict__ out) {
    int idx = blockIdx.x * blockDim.x + threadIdx.x;
    if (idx >= total) return;
    int e = idx & 15;
    int lane = (idx >> 4) & 31;
    int tile = idx >> 9;
    int nt = tile % NT;
    int kc = tile / NT;
    int nn = nt * 16 + (lane & 15);
    int kk = kc * 32 + 8 * (lane >> 4) + (e & 7) + 16 * (e >> 3);
    float v = (nn < Ncols) ? W[(size_t)kk * Ncols + nn] : 0.0f;
    out[idx] = (__bf16)v;
}

// ---------------- WMMA GEMM: out[n, NT*16] = A[n, KT*32] @ Bpack ------------
// A split across two row-major bf16 sources at kc == ksplit (wave-uniform).
template <int NT>
__global__ __launch_bounds__(256)
void wmma_gemm_k(const __bf16* __restrict__ A1, int lda1,
                 const __bf16* __restrict__ A2, int lda2, int ksplit,
                 const __bf16* __restrict__ Bpack,
                 const float* __restrict__ bias,
                 float* __restrict__ outF, __bf16* __restrict__ outB,
                 int ldout, int KT, int nrows, int do_relu) {
    const int wave = threadIdx.x >> 5;
    const int lane = threadIdx.x & 31;
    const int half = lane >> 4;
    const int m    = lane & 15;
    const int r0   = (blockIdx.x * (blockDim.x >> 5) + wave) * 16;
    if (r0 >= nrows) return;                      // wave-uniform: EXEC stays full
    const int row = r0 + m;

    v8f acc[NT];
#pragma unroll
    for (int nt = 0; nt < NT; nt++) acc[nt] = {};

    for (int kc = 0; kc < KT; kc++) {
        const __bf16* arow = (kc < ksplit)
            ? (A1 + (size_t)row * lda1 + kc * 32)
            : (A2 + (size_t)row * lda2 + (kc - ksplit) * 32);
        arow += 8 * half;
        v8bf lo = *(const v8bf*)(arow);           // K = kc*32 + 8*half + 0..7
        v8bf hi = *(const v8bf*)(arow + 16);      // K = kc*32 + 8*half + 16..23
        v16bf a;
#pragma unroll
        for (int i = 0; i < 8; i++) { a[i] = lo[i]; a[i + 8] = hi[i]; }

        const __bf16* bp = Bpack + ((size_t)(kc * NT) * 32 + lane) * 16;
#pragma unroll
        for (int nt = 0; nt < NT; nt++) {
            v16bf b = *(const v16bf*)(bp + (size_t)nt * 32 * 16);
            acc[nt] = __builtin_amdgcn_wmma_f32_16x16x32_bf16(
                false, a, false, b, (short)0, acc[nt], false, false);
        }
    }

#pragma unroll
    for (int nt = 0; nt < NT; nt++) {
        int col = nt * 16 + m;
        float bv = bias ? bias[col] : 0.0f;
#pragma unroll
        for (int v = 0; v < 8; v++) {
            int rr = r0 + v + 8 * half;           // C/D layout: row = v + 8*half
            float val = acc[nt][v] + bv;
            if (do_relu) val = fmaxf(val, 0.0f);
            if (outF) outF[(size_t)rr * ldout + col] = val;
            if (outB) outB[(size_t)rr * ldout + col] = (__bf16)val;
        }
    }
}

// ---------------- degree / normalization ----------------
__global__ void fill1_k(float* __restrict__ p, int n) {
    int i = blockIdx.x * blockDim.x + threadIdx.x;
    if (i < n) p[i] = 1.0f;
}
__global__ void deg_k(const int* __restrict__ dst, float* __restrict__ deg, int e) {
    int i = blockIdx.x * blockDim.x + threadIdx.x;
    if (i < e) atomicAdd(&deg[dst[i]], 1.0f);
}
__global__ void rsq_k(float* __restrict__ p, int n) {
    int i = blockIdx.x * blockDim.x + threadIdx.x;
    if (i < n) p[i] = rsqrtf(p[i]);
}

// ---------------- GCN aggregation ----------------
__global__ void agg_init_k(const float* __restrict__ lin, const float* __restrict__ bias,
                           const float* __restrict__ dis, float* __restrict__ agg,
                           int n, int C, int ldlin) {
    long long idx = (long long)blockIdx.x * blockDim.x + threadIdx.x;
    if (idx >= (long long)n * C) return;
    int i = (int)(idx / C), c = (int)(idx % C);
    float di = dis[i];
    agg[idx] = bias[c] + lin[(size_t)i * ldlin + c] * di * di;
}

__global__ void agg_edge_k(const int* __restrict__ src, const int* __restrict__ dst,
                           const float* __restrict__ lin, const float* __restrict__ dis,
                           float* __restrict__ agg, int e, int C, int ldlin,
                           int tpe, int cpt) {
    long long idx = (long long)blockIdx.x * blockDim.x + threadIdx.x;
    long long ee = idx / tpe;
    int part = (int)(idx % tpe);
    if (ee >= e) return;
    int s = src[ee], d = dst[ee];
    float w = dis[s] * dis[d];
    const float* lr = lin + (size_t)s * ldlin + part * cpt;
    float* ar = agg + (size_t)d * C + part * cpt;
    for (int c = 0; c < cpt; c++) atomicAdd(&ar[c], lr[c] * w);
}

__global__ void relu_b16_k(float* __restrict__ p, __bf16* __restrict__ pb, long long cnt) {
    long long i = (long long)blockIdx.x * blockDim.x + threadIdx.x;
    if (i >= cnt) return;
    float v = fmaxf(p[i], 0.0f);
    p[i] = v;
    pb[i] = (__bf16)v;
}

// ---------------- small 32x64 @ 64x64 GEMM ----------------
__global__ void small_gemm32_k(const float* __restrict__ A, const float* __restrict__ B,
                               float* __restrict__ O) {
    int idx = blockIdx.x * blockDim.x + threadIdx.x;
    if (idx >= 32 * 64) return;
    int r = idx >> 6, c = idx & 63;
    float s = 0.0f;
    for (int k = 0; k < 64; k++) s += A[r * 64 + k] * B[k * 64 + c];
    O[idx] = s;
}

// ---------------- memory-attention scores S[p=hh*32+q][k] ----------------
__global__ void scores_k(const float* __restrict__ K, const float* __restrict__ Qw,
                         float* __restrict__ S, int n) {
    __shared__ float q[2048];
    for (int i = threadIdx.x; i < 2048; i += blockDim.x) q[i] = Qw[i];
    __syncthreads();
    int k = blockIdx.x * blockDim.x + threadIdx.x;
    if (k >= n) return;
    float kr[64];
#pragma unroll
    for (int i = 0; i < 64; i++) kr[i] = K[(size_t)k * 64 + i];
#pragma unroll
    for (int hh = 0; hh < 4; hh++)
        for (int qq = 0; qq < 32; qq++) {
            float s = 0.0f;
#pragma unroll
            for (int d = 0; d < 16; d++) s += q[qq * 64 + hh * 16 + d] * kr[hh * 16 + d];
            S[(size_t)(hh * 32 + qq) * n + k] = 0.25f * s;
        }
}

// ---------------- top-5 threshold per score row ----------------
__global__ void top5_k(const float* __restrict__ S, float* __restrict__ thr,
                       float* __restrict__ rmax, int n) {
    int p = blockIdx.x;
    const float* row = S + (size_t)p * n;
    float t[5];
#pragma unroll
    for (int j = 0; j < 5; j++) t[j] = -3.0e38f;
    for (int k = threadIdx.x; k < n; k += blockDim.x) {
        float s = row[k];
        if (s > t[4]) {
            t[4] = s;
#pragma unroll
            for (int j = 4; j > 0; j--)
                if (t[j] > t[j - 1]) { float tm = t[j]; t[j] = t[j - 1]; t[j - 1] = tm; }
        }
    }
    __shared__ float sm[256 * 5];
#pragma unroll
    for (int j = 0; j < 5; j++) sm[threadIdx.x * 5 + j] = t[j];
    __syncthreads();
    if (threadIdx.x == 0) {
        float b[5];
#pragma unroll
        for (int j = 0; j < 5; j++) b[j] = -3.0e38f;
        for (int i = 0; i < 256 * 5; i++) {
            float s = sm[i];
            if (s > b[4]) {
                b[4] = s;
#pragma unroll
                for (int j = 4; j > 0; j--)
                    if (b[j] > b[j - 1]) { float tm = b[j]; b[j] = b[j - 1]; b[j - 1] = tm; }
            }
        }
        thr[p] = b[4];
        rmax[p] = b[0];
    }
}

// ---------------- masked softmax * V -> mem_att[32,64] ----------------
__global__ void mematt_k(const float* __restrict__ S, const float* __restrict__ V,
                         const float* __restrict__ thr, const float* __restrict__ rmax,
                         float* __restrict__ matt, int n) {
    int p = blockIdx.x;
    int hh = p >> 5, qq = p & 31;
    float th = thr[p], mx = rmax[p];
    float den = 0.0f, acc[16];
#pragma unroll
    for (int d = 0; d < 16; d++) acc[d] = 0.0f;
    for (int k = threadIdx.x; k < n; k += blockDim.x) {
        float s = S[(size_t)p * n + k];
        if (s >= th) {
            float w = __expf(s - mx);
            den += w;
            const float* vr = V + (size_t)k * 64 + hh * 16;
#pragma unroll
            for (int d = 0; d < 16; d++) acc[d] += w * vr[d];
        }
    }
    __shared__ float sm[256][17];
    sm[threadIdx.x][0] = den;
#pragma unroll
    for (int d = 0; d < 16; d++) sm[threadIdx.x][1 + d] = acc[d];
    __syncthreads();
    for (int off = 128; off > 0; off >>= 1) {
        if ((int)threadIdx.x < off)
            for (int j = 0; j < 17; j++) sm[threadIdx.x][j] += sm[threadIdx.x + off][j];
        __syncthreads();
    }
    if (threadIdx.x < 16)
        matt[qq * 64 + hh * 16 + threadIdx.x] = sm[0][1 + threadIdx.x] / sm[0][0];
}

// ---------------- broadcast attention (per node, 32 keys) -> bf16 heads ----
__global__ void battn_k(const float* __restrict__ Qb, const float* __restrict__ Kb,
                        const float* __restrict__ Vb, __bf16* __restrict__ outB, int n) {
    __shared__ float kb[2048], vb[2048];
    for (int i = threadIdx.x; i < 2048; i += blockDim.x) { kb[i] = Kb[i]; vb[i] = Vb[i]; }
    __syncthreads();
    long long g = (long long)blockIdx.x * blockDim.x + threadIdx.x;
    int node = (int)(g >> 2), hh = (int)(g & 3);
    if (node >= n) return;
    float q[16];
#pragma unroll
    for (int d = 0; d < 16; d++) q[d] = Qb[(size_t)node * 64 + hh * 16 + d];
    float s[32], mx = -3.0e38f;
#pragma unroll
    for (int j = 0; j < 32; j++) {
        float t = 0.0f;
#pragma unroll
        for (int d = 0; d < 16; d++) t += q[d] * kb[j * 64 + hh * 16 + d];
        t *= 0.25f;
        s[j] = t;
        mx = fmaxf(mx, t);
    }
    float den = 0.0f;
#pragma unroll
    for (int j = 0; j < 32; j++) { s[j] = __expf(s[j] - mx); den += s[j]; }
    float inv = 1.0f / den;
#pragma unroll
    for (int d = 0; d < 16; d++) {
        float o = 0.0f;
#pragma unroll
        for (int j = 0; j < 32; j++) o += s[j] * vb[j * 64 + hh * 16 + d];
        outB[(size_t)node * 64 + hh * 16 + d] = (__bf16)(o * inv);
    }
}

// ---------------- LayerNorm(X+Y) over 64, wave-per-row ----------------
__global__ void ln_k(const float* __restrict__ X, const float* __restrict__ Y,
                     const float* __restrict__ g, const float* __restrict__ b,
                     float* __restrict__ outF, __bf16* __restrict__ outB, int n) {
    int wave = threadIdx.x >> 5, lane = threadIdx.x & 31;
    int row = blockIdx.x * (blockDim.x >> 5) + wave;
    if (row >= n) return;
    size_t base = (size_t)row * 64;
    float v0 = X[base + lane], v1 = X[base + lane + 32];
    if (Y) { v0 += Y[base + lane]; v1 += Y[base + lane + 32]; }
    float s = v0 + v1;
    for (int off = 16; off; off >>= 1) s += __shfl_xor(s, off, 32);
    float mu = s * (1.0f / 64.0f);
    float d0 = v0 - mu, d1 = v1 - mu;
    float vv = d0 * d0 + d1 * d1;
    for (int off = 16; off; off >>= 1) vv += __shfl_xor(vv, off, 32);
    float rs = rsqrtf(vv * (1.0f / 64.0f) + 1e-5f);
    float r0 = d0 * rs * g[lane] + b[lane];
    float r1 = d1 * rs * g[lane + 32] + b[lane + 32];
    if (outF) { outF[base + lane] = r0; outF[base + lane + 32] = r1; }
    if (outB) { outB[base + lane] = (__bf16)r0; outB[base + lane + 32] = (__bf16)r1; }
}

// ---------------- z = 0.5*x_gw + LN(h) -> bf16 ----------------
__global__ void zcomb_k(const float* __restrict__ h, const float* __restrict__ xgw,
                        const float* __restrict__ g, const float* __restrict__ b,
                        __bf16* __restrict__ outB, int n) {
    int wave = threadIdx.x >> 5, lane = threadIdx.x & 31;
    int row = blockIdx.x * (blockDim.x >> 5) + wave;
    if (row >= n) return;
    size_t base = (size_t)row * 64;
    float v0 = h[base + lane], v1 = h[base + lane + 32];
    float s = v0 + v1;
    for (int off = 16; off; off >>= 1) s += __shfl_xor(s, off, 32);
    float mu = s * (1.0f / 64.0f);
    float d0 = v0 - mu, d1 = v1 - mu;
    float vv = d0 * d0 + d1 * d1;
    for (int off = 16; off; off >>= 1) vv += __shfl_xor(vv, off, 32);
    float rs = rsqrtf(vv * (1.0f / 64.0f) + 1e-5f);
    float z0 = 0.5f * xgw[base + lane]      + (d0 * rs * g[lane] + b[lane]);
    float z1 = 0.5f * xgw[base + lane + 32] + (d1 * rs * g[lane + 32] + b[lane + 32]);
    outB[base + lane] = (__bf16)z0;
    outB[base + lane + 32] = (__bf16)z1;
}

// ---------------- log_softmax over 40 ----------------
__global__ void lsm_k(const float* __restrict__ agg, float* __restrict__ out, int n) {
    int i = blockIdx.x * blockDim.x + threadIdx.x;
    if (i >= n) return;
    const float* a = agg + (size_t)i * 40;
    float mx = -3.0e38f;
#pragma unroll
    for (int c = 0; c < 40; c++) mx = fmaxf(mx, a[c]);
    float den = 0.0f;
#pragma unroll
    for (int c = 0; c < 40; c++) den += __expf(a[c] - mx);
    float l = logf(den);
    float* o = out + (size_t)i * 40;
#pragma unroll
    for (int c = 0; c < 40; c++) o[c] = a[c] - mx - l;
}

// ===========================================================================
extern "C" void kernel_launch(void* const* d_in, const int* in_sizes, int n_in,
                              void* d_out, int out_size, void* d_ws, size_t ws_size,
                              hipStream_t stream) {
    const float* x      = (const float*)d_in[0];
    const float* pe     = (const float*)d_in[1];
    const float* memory = (const float*)d_in[2];
    const float* gcn_W1 = (const float*)d_in[3];
    const float* gcn_b1 = (const float*)d_in[4];
    const float* gcn_W2 = (const float*)d_in[5];
    const float* gcn_b2 = (const float*)d_in[6];
    const float* Wqw = (const float*)d_in[7];
    const float* Wkw = (const float*)d_in[8];
    const float* Wvw = (const float*)d_in[9];
    const float* Wow = (const float*)d_in[10];
    const float* Wqb = (const float*)d_in[11];
    const float* Wkb = (const float*)d_in[12];
    const float* Wvb = (const float*)d_in[13];
    const float* Wob = (const float*)d_in[14];
    const float* ln1_g = (const float*)d_in[15];
    const float* ln1_b = (const float*)d_in[16];
    const float* ffn_W1 = (const float*)d_in[17];
    const float* ffn_b1 = (const float*)d_in[18];
    const float* ffn_W2 = (const float*)d_in[19];
    const float* ffn_b2 = (const float*)d_in[20];
    const float* ln2_g = (const float*)d_in[21];
    const float* ln2_b = (const float*)d_in[22];
    const float* lnm_g = (const float*)d_in[23];
    const float* lnm_b = (const float*)d_in[24];
    const int*   ei    = (const int*)d_in[25];

    const int n = in_sizes[0] / 128;     // 100000
    const int E = in_sizes[25] / 2;      // 3200000
    const int* esrc = ei;
    const int* edst = ei + E;
    float* out = (float*)d_out;

    char* ws = (char*)d_ws;
    __bf16* XC   = (__bf16*)(ws + 0);              // [n,192] bf16
    __bf16* HB   = (__bf16*)(ws + 38400000);       // [n,64]  bf16 (h / attnH / t / z)
    float*  P0   = (float*)(ws + 51200000);        // [n,64]  f32
    float*  P1   = (float*)(ws + 76800000);        // [n,64]  f32 (h)
    float*  P2   = (float*)(ws + 102400000);       // [n,64]/[n,48] f32
    float*  SB   = (float*)(ws + 128000000);       // scores [128,n] f32
    __bf16* MIDB = (__bf16*)(ws + 128000000);      // ffn mid [n,128] bf16 (reuse)
    float*  AG2  = (float*)(ws + 153600000);       // [n,40] f32
    char*   SM   = ws + 179200000;
    float* dis  = (float*)(SM);
    float* thr  = (float*)(SM + 400000);
    float* rmax = (float*)(SM + 400512);
    float* Qw   = (float*)(SM + 401024);
    float* Kb   = (float*)(SM + 409216);
    float* Vb   = (float*)(SM + 417408);
    float* MATT = (float*)(SM + 425600);
    float* MNEW = (float*)(SM + 433792);
    __bf16* PK  = (__bf16*)(SM + 458752);
    __bf16* pW1  = PK;            // 6*4*512 = 12288
    __bf16* pWk  = PK + 12288;    // 2*4*512 = 4096
    __bf16* pWv  = PK + 16384;
    __bf16* pWqb = PK + 20480;
    __bf16* pWob = PK + 24576;
    __bf16* pF1  = PK + 28672;    // 2*8*512 = 8192
    __bf16* pF2  = PK + 36864;    // 4*4*512 = 8192
    __bf16* pW2  = PK + 45056;    // 8*3*512 = 12288

    // ---- prep: xc bf16, weight packs ----
    pack_xc_k<<<DIV_UP((long long)n * 192, 256), 256, 0, stream>>>(x, pe, XC, n);
    pack_w_k<<<DIV_UP(12288, 256), 256, 0, stream>>>(gcn_W1, 64, 4, 12288, pW1);
    pack_w_k<<<DIV_UP(4096, 256), 256, 0, stream>>>(Wkw, 64, 4, 4096, pWk);
    pack_w_k<<<DIV_UP(4096, 256), 256, 0, stream>>>(Wvw, 64, 4, 4096, pWv);
    pack_w_k<<<DIV_UP(4096, 256), 256, 0, stream>>>(Wqb, 64, 4, 4096, pWqb);
    pack_w_k<<<DIV_UP(4096, 256), 256, 0, stream>>>(Wob, 64, 4, 4096, pWob);
    pack_w_k<<<DIV_UP(8192, 256), 256, 0, stream>>>(ffn_W1, 128, 8, 8192, pF1);
    pack_w_k<<<DIV_UP(8192, 256), 256, 0, stream>>>(ffn_W2, 64, 4, 8192, pF2);
    pack_w_k<<<DIV_UP(12288, 256), 256, 0, stream>>>(gcn_W2, 40, 3, 12288, pW2);

    // ---- degree normalization ----
    fill1_k<<<DIV_UP(n, 256), 256, 0, stream>>>(dis, n);
    deg_k<<<DIV_UP(E, 256), 256, 0, stream>>>(edst, dis, E);
    rsq_k<<<DIV_UP(n, 256), 256, 0, stream>>>(dis, n);

    const int ggrid = DIV_UP(n, 128);   // 8 waves * 16 rows per block

    // ---- GCN1: hlin = xc @ W1 ; aggregate ; relu -> h (P1 f32, HB bf16) ----
    wmma_gemm_k<4><<<ggrid, 256, 0, stream>>>(XC, 192, XC, 192, 6, pW1, nullptr,
                                              P0, nullptr, 64, 6, n, 0);
    agg_init_k<<<DIV_UP((long long)n * 64, 256), 256, 0, stream>>>(P0, gcn_b1, dis, P1, n, 64, 64);
    agg_edge_k<<<DIV_UP((long long)E * 4, 256), 256, 0, stream>>>(esrc, edst, P0, dis, P1,
                                                                  E, 64, 64, 4, 16);
    relu_b16_k<<<DIV_UP((long long)n * 64, 256), 256, 0, stream>>>(P1, HB, (long long)n * 64);

    // ---- memory attention (write phase, top-5) ----
    small_gemm32_k<<<8, 256, 0, stream>>>(memory, Wqw, Qw);                       // Q = mem@Wq
    wmma_gemm_k<4><<<ggrid, 256, 0, stream>>>(HB, 64, HB, 64, 2, pWk, nullptr,
                                              P0, nullptr, 64, 2, n, 0);          // K = h@Wk
    wmma_gemm_k<4><<<ggrid, 256, 0, stream>>>(HB, 64, HB, 64, 2, pWv, nullptr,
                                              P2, nullptr, 64, 2, n, 0);          // V = h@Wv
    scores_k<<<DIV_UP(n, 256), 256, 0, stream>>>(P0, Qw, SB, n);
    top5_k<<<128, 256, 0, stream>>>(SB, thr, rmax, n);
    mematt_k<<<128, 256, 0, stream>>>(SB, P2, thr, rmax, MATT, n);
    small_gemm32_k<<<8, 256, 0, stream>>>(MATT, Wow, MNEW);                       // mem_new
    small_gemm32_k<<<8, 256, 0, stream>>>(MNEW, Wkb, Kb);
    small_gemm32_k<<<8, 256, 0, stream>>>(MNEW, Wvb, Vb);

    // ---- broadcast attention ----
    wmma_gemm_k<4><<<ggrid, 256, 0, stream>>>(HB, 64, HB, 64, 2, pWqb, nullptr,
                                              P0, nullptr, 64, 2, n, 0);          // Qb = h@Wqb
    battn_k<<<DIV_UP((long long)n * 4, 256), 256, 0, stream>>>(P0, Kb, Vb, HB, n); // heads->HB bf16
    wmma_gemm_k<4><<<ggrid, 256, 0, stream>>>(HB, 64, HB, 64, 2, pWob, nullptr,
                                              P0, nullptr, 64, 2, n, 0);          // attn = heads@Wob

    // ---- t = LN(h + attn) -> P2 f32 + HB bf16 ----
    ln_k<<<DIV_UP(n, 8), 256, 0, stream>>>(P1, P0, ln1_g, ln1_b, P2, HB, n);

    // ---- FFN: mid = relu(t@W1+b1) bf16 ; f = mid@W2+b2 -> P0 ----
    wmma_gemm_k<8><<<ggrid, 256, 0, stream>>>(HB, 64, HB, 64, 2, pF1, ffn_b1,
                                              nullptr, MIDB, 128, 2, n, 1);
    wmma_gemm_k<4><<<ggrid, 256, 0, stream>>>(MIDB, 128, MIDB, 128, 4, pF2, ffn_b2,
                                              P0, nullptr, 64, 4, n, 0);

    // ---- x_gw = LN(t + f) -> P0 ; z = 0.5*x_gw + LN(h) -> HB bf16 ----
    ln_k<<<DIV_UP(n, 8), 256, 0, stream>>>(P2, P0, ln2_g, ln2_b, P0, nullptr, n);
    zcomb_k<<<DIV_UP(n, 8), 256, 0, stream>>>(P1, P0, lnm_g, lnm_b, HB, n);

    // ---- GCN2: zlin = [z | xc] @ W2 (k-split A) ; aggregate ; log_softmax ----
    wmma_gemm_k<3><<<ggrid, 256, 0, stream>>>(HB, 64, XC, 192, 2, pW2, nullptr,
                                              P2, nullptr, 48, 8, n, 0);
    agg_init_k<<<DIV_UP((long long)n * 40, 256), 256, 0, stream>>>(P2, gcn_b2, dis, AG2, n, 40, 48);
    agg_edge_k<<<DIV_UP((long long)E * 4, 256), 256, 0, stream>>>(esrc, edst, P2, dis, AG2,
                                                                  E, 40, 48, 4, 10);
    lsm_k<<<DIV_UP(n, 256), 256, 0, stream>>>(AG2, out, n);

    (void)n_in; (void)out_size; (void)ws_size;
}
